// CustomMHAlayer_58171037057224
// MI455X (gfx1250) — matrix operand order repaired and checked by reading, not compiled, not run on hardware
//
#include <hip/hip_runtime.h>
#include <math.h>

typedef unsigned short u16;
typedef __attribute__((ext_vector_type(16))) __bf16          v16bf;
typedef __attribute__((ext_vector_type(8)))  float           v8f;
typedef __attribute__((ext_vector_type(16))) unsigned short  v16u;
typedef __attribute__((ext_vector_type(8)))  unsigned short  v8u;

#define E_DIM 512
#define HEADS 16
#define DH    32

// ---------------- bf16 helpers ----------------
__device__ __forceinline__ u16 f2bf(float f) {
  unsigned u = __float_as_uint(f);
  unsigned r = u + 0x7FFFu + ((u >> 16) & 1u);   // round-to-nearest-even
  return (u16)(r >> 16);
}
__device__ __forceinline__ float bf2f(u16 h) {
  return __uint_as_float(((unsigned)h) << 16);
}
__device__ __forceinline__ v16bf mk_frag(v8u lo, v8u hi) {
  v16u t;
#pragma unroll
  for (int i = 0; i < 8; ++i) { t[i] = lo[i]; t[i + 8] = hi[i]; }
  return __builtin_bit_cast(v16bf, t);
}
__device__ __forceinline__ v8f wmma_bf16(v16bf a, v16bf b, v8f c) {
  // D(16x16 f32) = A(16x32 bf16) x B(32x16 bf16) + C
  return __builtin_amdgcn_wmma_f32_16x16x32_bf16(false, a, false, b, (short)0, c,
                                                 false, false);
}

// ---------------- CDNA5 async global->LDS copy (ASYNCcnt path) ----------------
// Each lane copies 16B: LDS[lds_off] = MEM[gaddr]. Tracked with ASYNCcnt.
__device__ __forceinline__ void async_load_b128(unsigned lds_off, const void* gaddr) {
  asm volatile("global_load_async_to_lds_b128 %0, %1, off"
               :: "v"(lds_off), "v"(gaddr)
               : "memory");
}
__device__ __forceinline__ void wait_asynccnt0() {
  asm volatile("s_wait_asynccnt 0x0" ::: "memory");
}

// ---------------- f32 -> bf16 conversion ----------------
__global__ void cvt_f32_bf16(const float* __restrict__ src, u16* __restrict__ dst, int n) {
  int i = (blockIdx.x * 256 + threadIdx.x) * 4;
  if (i >= n) return;
  float4 v = *(const float4*)(src + i);
  uint2 pk;
  pk.x = (unsigned)f2bf(v.x) | ((unsigned)f2bf(v.y) << 16);
  pk.y = (unsigned)f2bf(v.z) | ((unsigned)f2bf(v.w) << 16);
  *(uint2*)(dst + i) = pk;
}

// ---------------- WMMA GEMM: Y = X @ W^T + bias ----------------
// X: (M x K) bf16 row-major (ldx), W: (N x K) bf16 row-major (ldw).
// Wave computes a 32x32 tile (2x2 WMMA register blocking): 4 WMMAs per 8 b128
// loads per K-step of 32. Block = 8 waves (2M x 4N) -> 64x128 tile.
// Output row remap (for batch-interleaved concatenation):
//   yr = row + (row / rows_per_batch) * y_pad_rows + y_row_off
template <bool OUT_BF16>
__global__ __launch_bounds__(256) void gemm_bias(
    int M, int N, int K, const u16* __restrict__ X, int ldx,
    const u16* __restrict__ W, int ldw, const float* __restrict__ bias,
    void* __restrict__ Yv, int ldy, int rows_per_batch, int y_pad_rows, int y_row_off) {
  const int lane = threadIdx.x & 31;
  const int wave = threadIdx.x >> 5;                 // 8 waves: 2(M) x 4(N)
  const int l15 = lane & 15, hi = lane >> 4;
  const int mtile = blockIdx.y * 64 + (wave >> 2) * 32;
  const int ntile = blockIdx.x * 128 + (wave & 3) * 32;
  (void)M; (void)N;

  const u16* xr0 = X + (size_t)(mtile + l15) * ldx;        // A rows (lane = M)
  const u16* xr1 = X + (size_t)(mtile + 16 + l15) * ldx;
  const u16* wr0 = W + (size_t)(ntile + l15) * ldw;        // B cols (lane = N)
  const u16* wr1 = W + (size_t)(ntile + 16 + l15) * ldw;
  const float bv0 = bias ? bias[ntile + l15] : 0.f;
  const float bv1 = bias ? bias[ntile + 16 + l15] : 0.f;
  v8f c00, c01, c10, c11;
#pragma unroll
  for (int i = 0; i < 8; ++i) { c00[i] = bv0; c01[i] = bv1; c10[i] = bv0; c11[i] = bv1; }

  for (int kk = 0; kk < K; kk += 32) {
    const int ka = kk + hi * 16;                 // A K-chunk: 16 contiguous
    v16bf a0 = mk_frag(*(const v8u*)(xr0 + ka), *(const v8u*)(xr0 + ka + 8));
    v16bf a1 = mk_frag(*(const v8u*)(xr1 + ka), *(const v8u*)(xr1 + ka + 8));
    const int kb = kk + hi * 16;                 // B K-chunk: 16 contiguous
    v16bf b0 = __builtin_bit_cast(v16bf, *(const v16u*)(wr0 + kb));
    v16bf b1 = __builtin_bit_cast(v16bf, *(const v16u*)(wr1 + kb));
    c00 = wmma_bf16(a0, b0, c00);
    c01 = wmma_bf16(a0, b1, c01);
    c10 = wmma_bf16(a1, b0, c10);
    c11 = wmma_bf16(a1, b1, c11);
  }

  auto store_tile = [&](const v8f& c, int mb, int nb) {
    const int col = nb + l15;
#pragma unroll
    for (int r = 0; r < 8; ++r) {
      int row = mb + r + 8 * hi;
      int yr = row + (row / rows_per_batch) * y_pad_rows + y_row_off;
      if (OUT_BF16) ((u16*)Yv)[(size_t)yr * ldy + col] = f2bf(c[r]);
      else          ((float*)Yv)[(size_t)yr * ldy + col] = c[r];
    }
  };
  store_tile(c00, mtile, ntile);
  store_tile(c01, mtile, ntile + 16);
  store_tile(c10, mtile + 16, ntile);
  store_tile(c11, mtile + 16, ntile + 16);
}

// ---------------- Fused flash attention (WMMA, online softmax) ----------------
// Q: (B*S x E) bf16. K,V: (B*S x E) for self, (B*2S x E) for cross (pcpt rows
// first, then gen rows, per batch). CROSS adds the diagonal gen key S+i.
template <bool CROSS>
__global__ __launch_bounds__(128) void attn_fused(
    const u16* __restrict__ Q, const u16* __restrict__ K, const u16* __restrict__ V,
    u16* __restrict__ O, int S) {
  const int lane = threadIdx.x & 31;
  const int wave = threadIdx.x >> 5;   // 4 waves, 16 queries each
  const int l15 = lane & 15, hi = lane >> 4;
  const int b = blockIdx.z, h = blockIdx.y;
  const int qw = blockIdx.x * 64 + wave * 16;              // local query base
  const size_t qrow0  = (size_t)b * S + qw;
  const size_t kvrow0 = (size_t)b * (CROSS ? 2 * S : S);

  __shared__ u16   sK[32][40];        // [key][dim], padded vs bank conflicts
  __shared__ u16   sV[32][40];        // [dim][key]  (transposed stage)
  __shared__ u16   sP[4][16][32];     // per-wave P transpose buffer
  __shared__ float sD[4][16];         // per-wave diagonal-score broadcast

  // Q fragment (A layout: lane = row, two contiguous 8-elem K-chunks)
  const u16* qp = Q + (qrow0 + l15) * E_DIM + h * DH;
  const v16bf qa = mk_frag(*(const v8u*)(qp + hi * 16), *(const v8u*)(qp + hi * 16 + 8));

  const float scale = 0.17677669529663688f;  // 1/sqrt(32)
  float m_[8], l_[8];
  v8f o0 = {}, o1 = {};
#pragma unroll
  for (int r = 0; r < 8; ++r) { m_[r] = -3.0e38f; l_[r] = 0.f; }

  const int tid = threadIdx.x;
  const int ldkey = tid >> 2;          // 0..31
  const int lddim = (tid & 3) * 8;     // 0,8,16,24

  for (int kb = 0; kb < S; kb += 32) {
    __syncthreads();                   // protect LDS tiles from prior-iter readers
    {
      // K tile: async global->LDS (ASYNCcnt), verbatim row-major copy
      const u16* kp = K + (kvrow0 + kb + ldkey) * E_DIM + h * DH + lddim;
      async_load_b128((unsigned)(size_t)&sK[ldkey][lddim], kp);
      // V tile: register transpose into dim-major LDS
      const u16* vp = V + (kvrow0 + kb + ldkey) * E_DIM + h * DH + lddim;
      v8u vv = *(const v8u*)vp;
#pragma unroll
      for (int j = 0; j < 8; ++j) sV[lddim + j][ldkey] = vv[j];
    }
    wait_asynccnt0();                  // this wave's async LDS writes landed
    __syncthreads();

    // scores: two 16x16 tiles over 32 keys
    const u16* p0 = &sK[l15][hi * 16];
    const u16* p1 = &sK[16 + l15][hi * 16];
    v16bf bk0 = mk_frag(*(const v8u*)p0, *(const v8u*)(p0 + 8));
    v16bf bk1 = mk_frag(*(const v8u*)p1, *(const v8u*)(p1 + 8));
    v8f z = {};
    v8f c0 = wmma_bf16(qa, bk0, z);
    v8f c1 = wmma_bf16(qa, bk1, z);

#pragma unroll
    for (int r = 0; r < 8; ++r) {
      float s0 = c0[r] * scale, s1 = c1[r] * scale;
      float mx = fmaxf(s0, s1);
#pragma unroll
      for (int d = 1; d < 16; d <<= 1) mx = fmaxf(mx, __shfl_xor(mx, d));
      float mn = fmaxf(m_[r], mx);
      float al = __expf(m_[r] - mn);
      float e0 = __expf(s0 - mn), e1 = __expf(s1 - mn);
      float sm = e0 + e1;
#pragma unroll
      for (int d = 1; d < 16; d <<= 1) sm += __shfl_xor(sm, d);
      l_[r] = l_[r] * al + sm;
      m_[r] = mn;
      o0[r] *= al; o1[r] *= al;
      int mrow = r + 8 * hi;
      sP[wave][mrow][l15] = f2bf(e0);
      sP[wave][mrow][16 + l15] = f2bf(e1);
    }

    // P transpose read (wave-local; LDS ops are in-order within a wave)
    const u16* pp = &sP[wave][l15][hi * 16];
    v16bf pa = mk_frag(*(const v8u*)pp, *(const v8u*)(pp + 8));
    const u16* v0p = &sV[l15][hi * 16];
    const u16* v1p = &sV[16 + l15][hi * 16];
    v16bf bv0 = mk_frag(*(const v8u*)v0p, *(const v8u*)(v0p + 8));
    v16bf bv1 = mk_frag(*(const v8u*)v1p, *(const v8u*)(v1p + 8));
    o0 = wmma_bf16(pa, bv0, o0);       // dims 0..15
    o1 = wmma_bf16(pa, bv1, o1);       // dims 16..31
  }

  if (CROSS) {
    // diagonal gen key: query i also attends key S+i of its batch
    {
      int q = qw + l15;
      const u16* qd = Q + ((size_t)b * S + q) * E_DIM + h * DH;
      const u16* kd = K + (kvrow0 + S + q) * E_DIM + h * DH;
      float acc = 0.f;
#pragma unroll
      for (int i = 0; i < DH; ++i) acc += bf2f(qd[i]) * bf2f(kd[i]);
      if (hi == 0) sD[wave][l15] = acc * scale;
    }
#pragma unroll
    for (int r = 0; r < 8; ++r) {
      int mrow = r + 8 * hi;
      float e = sD[wave][mrow];
      float mn = fmaxf(m_[r], e);
      float al = __expf(m_[r] - mn);
      float p = __expf(e - mn);
      l_[r] = l_[r] * al + p;
      m_[r] = mn;
      const u16* vd = V + (kvrow0 + S + qw + mrow) * E_DIM + h * DH;
      o0[r] = o0[r] * al + p * bf2f(vd[l15]);
      o1[r] = o1[r] * al + p * bf2f(vd[16 + l15]);
    }
  }

#pragma unroll
  for (int r = 0; r < 8; ++r) {
    float inv = 1.f / l_[r];
    int row = qw + r + 8 * hi;
    u16* op = O + ((size_t)b * S + row) * E_DIM + h * DH;
    op[l15] = f2bf(o0[r] * inv);
    op[16 + l15] = f2bf(o1[r] * inv);
  }
}

// ---------------- host orchestration ----------------
extern "C" void kernel_launch(void* const* d_in, const int* in_sizes, int n_in,
                              void* d_out, int out_size, void* d_ws, size_t ws_size,
                              hipStream_t stream) {
  (void)in_sizes; (void)n_in; (void)out_size; (void)ws_size;
  const int B = 2, S = 1536, E = 512;
  const int M = B * S;       // 3072
  const int N3 = 3 * E;      // 1536

  const float* x_p   = (const float*)d_in[0];
  const float* x_g   = (const float*)d_in[1];
  const float* wqkv  = (const float*)d_in[4];
  const float* bqkv  = (const float*)d_in[5];
  const float* wsin  = (const float*)d_in[6];
  const float* bsin  = (const float*)d_in[7];
  const float* wsout = (const float*)d_in[8];
  const float* bsout = (const float*)d_in[9];
  const float* wcin  = (const float*)d_in[10];
  const float* bcin  = (const float*)d_in[11];
  const float* wcout = (const float*)d_in[12];
  const float* bcout = (const float*)d_in[13];
  const float* wop   = (const float*)d_in[14];
  const float* bop   = (const float*)d_in[15];
  float* out = (float*)d_out;

  char* ws = (char*)d_ws;
  size_t off = 0;
  auto carve = [&](size_t elems) -> u16* {
    off = (off + 255) & ~(size_t)255;
    u16* p = (u16*)(ws + off);
    off += elems * sizeof(u16);
    return p;
  };
  u16* bXp    = carve((size_t)M * E);
  u16* bXg    = carve((size_t)M * E);
  u16* bWqkv  = carve((size_t)N3 * E);
  u16* bWsin  = carve((size_t)N3 * E);
  u16* bWsout = carve((size_t)E * E);
  u16* bWop   = carve((size_t)E * E);
  u16* bWcin  = carve((size_t)N3 * E);
  u16* bWcout = carve((size_t)E * E);
  u16* bPqkv  = carve((size_t)M * N3);
  u16* bGqkv  = carve((size_t)M * N3);
  u16* bQs    = carve((size_t)M * E);
  u16* bKs    = carve((size_t)M * E);
  u16* bVs    = carve((size_t)M * E);
  u16* bCtx1  = carve((size_t)M * E);
  u16* bT1    = carve((size_t)M * E);
  u16* bQc    = carve((size_t)M * E);
  u16* bKc    = carve((size_t)2 * M * E);
  u16* bVc    = carve((size_t)2 * M * E);
  u16* bCtx2  = carve((size_t)M * E);

  auto cvt = [&](const float* s, u16* d, size_t n) {
    cvt_f32_bf16<<<dim3((unsigned)((n / 4 + 255) / 256)), dim3(256), 0, stream>>>(s, d, (int)n);
  };
  cvt(x_p, bXp, (size_t)M * E);
  cvt(x_g, bXg, (size_t)M * E);
  cvt(wqkv, bWqkv, (size_t)N3 * E);
  cvt(wsin, bWsin, (size_t)N3 * E);
  cvt(wsout, bWsout, (size_t)E * E);
  cvt(wop, bWop, (size_t)E * E);
  cvt(wcin, bWcin, (size_t)N3 * E);
  cvt(wcout, bWcout, (size_t)E * E);

  dim3 blk(256);
  auto gb = [&](int m, int n, int k, const u16* X, int ldx, const u16* W,
                const float* bias, u16* Y, int ldy, int rpb, int pad, int roff) {
    gemm_bias<true><<<dim3(n / 128, m / 64), blk, 0, stream>>>(
        m, n, k, X, ldx, W, E, bias, (void*)Y, ldy, rpb, pad, roff);
  };

  // fused QKV projections (shared weights)
  gb(M, N3, E, bXp, E, bWqkv, bqkv, bPqkv, N3, M, 0, 0);
  gb(M, N3, E, bXg, E, bWqkv, bqkv, bGqkv, N3, M, 0, 0);
  // self-attention in_proj (per-slice weights applied to q/k/v slices)
  gb(M, E, E, bPqkv + 0,     N3, bWsin + 0,                 bsin + 0,     bQs, E, M, 0, 0);
  gb(M, E, E, bPqkv + E,     N3, bWsin + (size_t)E * E,     bsin + E,     bKs, E, M, 0, 0);
  gb(M, E, E, bPqkv + 2 * E, N3, bWsin + (size_t)2 * E * E, bsin + 2 * E, bVs, E, M, 0, 0);
  // fused self flash-attention
  attn_fused<false><<<dim3(S / 64, HEADS, B), dim3(128), 0, stream>>>(bQs, bKs, bVs, bCtx1, S);
  // self out_proj, then layer outproj -> d_out (f32)
  gb(M, E, E, bCtx1, E, bWsout, bsout, bT1, E, M, 0, 0);
  gemm_bias<false><<<dim3(E / 128, M / 64), blk, 0, stream>>>(
      M, E, E, bT1, E, bWop, E, bop, (void*)out, E, M, 0, 0);

  // cross in_proj; K/V written batch-interleaved: [b: pcpt S rows | gen S rows]
  gb(M, E, E, bGqkv + 0,     N3, bWcin + 0,                 bcin + 0,     bQc, E, M, 0, 0);
  gb(M, E, E, bPqkv + E,     N3, bWcin + (size_t)E * E,     bcin + E,     bKc, E, S, S, 0);
  gb(M, E, E, bGqkv + E,     N3, bWcin + (size_t)E * E,     bcin + E,     bKc, E, S, S, S);
  gb(M, E, E, bPqkv + 2 * E, N3, bWcin + (size_t)2 * E * E, bcin + 2 * E, bVc, E, S, S, 0);
  gb(M, E, E, bGqkv + 2 * E, N3, bWcin + (size_t)2 * E * E, bcin + 2 * E, bVc, E, S, S, S);
  // fused cross flash-attention (pcpt keys + diagonal gen key)
  attn_fused<true><<<dim3(S / 64, HEADS, B), dim3(128), 0, stream>>>(bQc, bKc, bVc, bCtx2, S);
  // cross out_proj -> d_out second half (f32); no layer outproj per reference
  gemm_bias<false><<<dim3(E / 128, M / 64), blk, 0, stream>>>(
      M, E, E, bCtx2, E, bWcout, E, bcout, (void*)(out + (size_t)M * E), E, M, 0, 0);
}
